// LAN_4398046511408
// MI455X (gfx1250) — compile-verified, matrix operand order checked by reading, not run on hardware
//
#include <hip/hip_runtime.h>

typedef __attribute__((ext_vector_type(16))) _Float16 v16h;
typedef __attribute__((ext_vector_type(8)))  float    v8f;

// K index inside a 16x32 (f16) WMMA A/B fragment for element i (0..15) of the
// per-lane v16h, given lane-half hi = lane/16.  (CDNA5 ISA 7.12.2)
__device__ __forceinline__ int kmap(int i, int hi) {
    return (((i >> 1) & 3) << 1) + (i & 1) + (hi << 3) + ((i >> 3) << 4);
}

__device__ __forceinline__ float sigmoid_f(float x) {
    return 1.0f / (1.0f + __expf(-x));
}

// attn logit for one (i,j) pair from folded per-token scalars.
// aq = (sq_phi, sq_ta, sq_tb, sq_tau) (q-side bias folded in)
// ak = (sk_phi, sk_ta, sk_tb, sk_tau) (k-side bias + const folded in)
__device__ __forceinline__ float lan_logit(const float4& aq, const float* ak) {
    float sphi = aq.x + ak[0];
    float sta  = aq.y + ak[1];
    float stb  = aq.z + ak[2];
    float stau = aq.w + ak[3];
    float phi  = sigmoid_f(sphi);
    float ti   = sigmoid_f(sta + stb);              // T_SCALAR == 1.0
    float sp   = fmaxf(stau, 0.0f) + __logf(1.0f + __expf(-fabsf(stau)));
    float tau  = sp + 1e-6f;
    return phi / tau * (1.0f - __expf(-tau * ti));
}

// ---------------------------------------------------------------------------
// Kernel A: fold the two-layer phi/tau stacks and Wta/Wtb into per-head
// 4-scalar projections of Wq / Wk:  WqF[512][32], WkF[512][32] (+ biases).
// col = h*4 + s,  s: 0=phi 1=ta 2=tb 3=tau
// ---------------------------------------------------------------------------
__global__ __launch_bounds__(256) void lan_fold(
    const float* __restrict__ Wq, const float* __restrict__ bq,
    const float* __restrict__ Wk, const float* __restrict__ bk,
    const float* __restrict__ Wphi_in,  const float* __restrict__ bphi_in,
    const float* __restrict__ Wphi_out, const float* __restrict__ bphi_out,
    const float* __restrict__ Wta, const float* __restrict__ bta,
    const float* __restrict__ Wtb, const float* __restrict__ btb,
    const float* __restrict__ Wtau_in,  const float* __restrict__ btau_in,
    const float* __restrict__ Wtau_out, const float* __restrict__ btau_out,
    float* __restrict__ WqF, float* __restrict__ WkF,
    float* __restrict__ bqF, float* __restrict__ bkF)
{
    __shared__ float wsq[4 * 64];   // q-side fold vectors, wsq[s*64+d]
    __shared__ float wsk[4 * 64];   // k-side fold vectors
    __shared__ float cons[4];       // per-scalar additive constants
    const int tid = threadIdx.x;

    if (tid < 128) {
        float sphi = 0.0f;
        for (int t = 0; t < 64; ++t) sphi += Wphi_in[tid * 64 + t] * Wphi_out[t];
        float stau = 0.0f;
        for (int t = 0; t < 32; ++t) stau += Wtau_in[tid * 32 + t] * Wtau_out[t];
        const int d = tid & 63;
        float* W = (tid >= 64) ? wsk : wsq;
        W[0 * 64 + d] = sphi;
        W[1 * 64 + d] = Wta[tid];
        W[2 * 64 + d] = Wtb[tid];
        W[3 * 64 + d] = stau;
    } else if (tid == 128) {
        float s = 0.0f;
        for (int t = 0; t < 64; ++t) s += bphi_in[t] * Wphi_out[t];
        cons[0] = s + bphi_out[0];
        cons[1] = bta[0];
        cons[2] = btb[0];
        float s2 = 0.0f;
        for (int t = 0; t < 32; ++t) s2 += btau_in[t] * Wtau_out[t];
        cons[3] = s2 + btau_out[0];
    }
    __syncthreads();

    for (int idx = tid; idx < 512 * 32; idx += 256) {
        const int c = idx >> 5, col = idx & 31, h = col >> 2, s = col & 3;
        const float* wq = &Wq[c * 512 + h * 64];
        const float* wk = &Wk[c * 512 + h * 64];
        float aq = 0.0f, ak = 0.0f;
        for (int d = 0; d < 64; ++d) {
            aq += wq[d] * wsq[s * 64 + d];
            ak += wk[d] * wsk[s * 64 + d];
        }
        WqF[idx] = aq;
        WkF[idx] = ak;
    }
    if (tid < 32) {
        const int h = tid >> 2, s = tid & 3;
        float s1 = 0.0f, s2 = 0.0f;
        for (int d = 0; d < 64; ++d) {
            s1 += bq[h * 64 + d] * wsq[s * 64 + d];
            s2 += bk[h * 64 + d] * wsk[s * 64 + d];
        }
        bqF[tid] = s1;
        bkF[tid] = s2 + cons[s];
    }
}

// ---------------------------------------------------------------------------
// Kernel B: per-token scalar projections
//   qsc/ksc[(bh*1024 + t)*4 + s] = x[row] . WF[:, h*4+s] + bias
// One block per token row (4096 blocks x 64 threads: 0..31 -> q, 32..63 -> k).
// ---------------------------------------------------------------------------
__global__ __launch_bounds__(64) void lan_scal(
    const float* __restrict__ x,
    const float* __restrict__ WqF, const float* __restrict__ WkF,
    const float* __restrict__ bqF, const float* __restrict__ bkF,
    float* __restrict__ qsc, float* __restrict__ ksc)
{
    __shared__ __align__(16) float xr[512];
    const int tid = threadIdx.x;
    const int row = blockIdx.x;                 // b*1024 + t
    const float4* g = (const float4*)(x + (size_t)row * 512);
    float4* d4 = (float4*)xr;
    for (int i = tid; i < 128; i += 64) d4[i] = g[i];
    __syncthreads();

    const int col = tid & 31;
    const bool isq = (tid < 32);
    const float* WF = isq ? WqF : WkF;
    float s = isq ? bqF[col] : bkF[col];
    for (int c = 0; c < 512; ++c) s += xr[c] * WF[c * 32 + col];

    const int b = row >> 10, t = row & 1023, h = col >> 2, ss = col & 3;
    float* out = isq ? qsc : ksc;
    out[(((size_t)(b * 8 + h) * 1024) + t) * 4 + ss] = s;
}

// ---------------------------------------------------------------------------
// WMMA GEMM: OUT[M,N] = A[M,K] (f32) x W[K,N] (f32) + bias[N]
// f32 -> f16 conversion at fragment build, f32 accumulate via
// v_wmma_f32_16x16x32_f16.  Block = 256 threads (8 waves), macro-tile 128x64,
// each wave: 16 M-rows x 64 N-cols (4 accumulators).  Optional f32 and f16
// outputs (v-projection wants f16 for the attention WMMA consumer).
// ---------------------------------------------------------------------------
__global__ __launch_bounds__(256) void lan_gemm(
    const float* __restrict__ A, const float* __restrict__ W,
    const float* __restrict__ bias,
    float* __restrict__ outf, _Float16* __restrict__ outh,
    int Msz, int Nsz, int Ksz)
{
    __shared__ __align__(16) float la[128 * 32];   // A tile
    __shared__ __align__(16) float lw[32 * 64];    // W tile
    const int tid = threadIdx.x;
    const int w = tid >> 5, L = tid & 31, m = L & 15, hi = L >> 4;
    const int m0 = blockIdx.x * 128;
    const int n0 = blockIdx.y * 64;

    v8f acc[4] = {};

    for (int kc = 0; kc < Ksz; kc += 32) {
        __syncthreads();
        {   // stage A tile 128x32 (16 floats / thread)
            const int r = tid >> 1, ko = (tid & 1) * 16;
            const float4* g = (const float4*)(A + (size_t)(m0 + r) * Ksz + kc + ko);
            float4* d4 = (float4*)(la + r * 32 + ko);
            d4[0] = g[0]; d4[1] = g[1]; d4[2] = g[2]; d4[3] = g[3];
            if (kc + 32 < Ksz)   // hint next A tile into L2 (global_prefetch_b8)
                __builtin_prefetch(A + (size_t)(m0 + r) * Ksz + kc + 32 + ko, 0, 1);
        }
        {   // stage W tile 32x64 (8 floats / thread)
            const int idx = tid * 2;           // float4 index within tile
            const int r = idx >> 4, co = (idx & 15) * 4;
            const float4* g = (const float4*)(W + (size_t)(kc + r) * Nsz + n0 + co);
            float4* d4 = (float4*)(lw + r * 64 + co);
            d4[0] = g[0]; d4[1] = g[1];
        }
        __syncthreads();

        v16h af;
#pragma unroll
        for (int i = 0; i < 16; ++i)
            af[i] = (_Float16)la[(w * 16 + m) * 32 + kmap(i, hi)];
#pragma unroll
        for (int dt = 0; dt < 4; ++dt) {
            v16h bf;
#pragma unroll
            for (int i = 0; i < 16; ++i)
                bf[i] = (_Float16)lw[kmap(i, hi) * 64 + dt * 16 + m];
            acc[dt] = __builtin_amdgcn_wmma_f32_16x16x32_f16(
                false, af, false, bf, (short)0, acc[dt], false, false);
        }
    }

    const int grow0 = m0 + w * 16;
#pragma unroll
    for (int dt = 0; dt < 4; ++dt) {
        const int col = n0 + dt * 16 + m;
        const float bv = bias ? bias[col] : 0.0f;
#pragma unroll
        for (int r = 0; r < 8; ++r) {
            const float v = acc[dt][r] + bv;
            const size_t idx = (size_t)(grow0 + r + 8 * hi) * Nsz + col;
            if (outf) outf[idx] = v;
            if (outh) outh[idx] = (_Float16)v;
        }
    }
}

// ---------------------------------------------------------------------------
// Pack kernel: rearrange v (f16, [4096][512]) into pre-swizzled WMMA
// B-fragments so the attention inner loop does one coalesced 32-byte load per
// fragment instead of 16 LDS gathers + cvts.
//   vp[((bh*32 + jc)*4 + dt)*32 + lane] = v16h fragment for chunk jc, d-tile dt
// Grid (32 bh, 32 jc), 128 threads (wave w handles d-tile dt = w).
// ---------------------------------------------------------------------------
__global__ __launch_bounds__(128) void lan_pack(
    const _Float16* __restrict__ vh, v16h* __restrict__ vp)
{
    const int tid = threadIdx.x;
    const int dt = tid >> 5, L = tid & 31, m = L & 15, hi = L >> 4;
    const int bh = blockIdx.x, jc = blockIdx.y;
    const int b = bh >> 3, h = bh & 7;

    v16h frag;
#pragma unroll
    for (int i = 0; i < 16; ++i) {
        const int j = jc * 32 + kmap(i, hi);
        frag[i] = vh[(size_t)(b * 1024 + j) * 512 + h * 64 + dt * 16 + m];
    }
    vp[((size_t)(bh * 32 + jc) * 4 + dt) * 32 + L] = frag;
}

// ---------------------------------------------------------------------------
// Attention kernel.  Grid (32 bh, 32 q-blocks of 32 rows), 64 threads
// (2 independent waves).  Lane L's q row is M = L%16, matching the WMMA
// A-fragment layout, so probabilities are produced directly in-register.
// Pass 1 computes streaming softmax stats (lane pair split over even/odd j,
// combined with __shfl_xor 16) AND caches logits as f16 in LDS, so pass 2 is
// a single v_exp per element.  V arrives pre-packed as B-fragments: no LDS
// staging, no barriers, 4x v_wmma_f32_16x16x32_f16 per 32-j chunk.
// LDS: 16 KB k-scalars + 64 KB logit cache = 80 KB  (320 KB/WGP -> 4 blocks).
// ---------------------------------------------------------------------------
__global__ __launch_bounds__(64) void lan_attn(
    const float* __restrict__ qsc, const float* __restrict__ ksc,
    const v16h* __restrict__ vp, float* __restrict__ o)
{
    __shared__ __align__(16) float    ak[1024 * 4];      // k-side scalars
    __shared__ __align__(16) _Float16 lf[2 * 16 * 1024]; // per-wave logit cache

    const int tid = threadIdx.x;
    const int bh = blockIdx.x;                 // 0..31
    const int b = bh >> 3, h = bh & 7;
    const int w = tid >> 5, L = tid & 31, m = L & 15, hi = L >> 4;

    {   // stage all 1024x4 k-scalars for this (b,h)
        const float4* g = (const float4*)(ksc + (size_t)bh * 1024 * 4);
        float4* d4 = (float4*)ak;
        for (int i = tid; i < 1024; i += 64) d4[i] = g[i];
    }
    __syncthreads();

    const int qt0 = blockIdx.y * 32 + w * 16;
    const int row = qt0 + m;                   // this lane's q token
    const float4 aq = *(const float4*)(qsc + ((size_t)bh * 1024 + row) * 4);
    _Float16* lfr = &lf[(w * 16 + m) * 1024]; // this lane's logit row

    // ---- pass 1: per-row max & sum(exp), cache logits ---------------------
    float mx = -3.0e38f, l = 0.0f;
    for (int j = hi; j < 1024; j += 2) {
        const float lg = lan_logit(aq, &ak[j * 4]);
        lfr[j] = (_Float16)lg;
        if (lg > mx) { l = l * __expf(mx - lg) + 1.0f; mx = lg; }
        else         { l += __expf(lg - mx); }
    }
    {
        const float mx2 = __shfl_xor(mx, 16, 32);
        const float l2  = __shfl_xor(l, 16, 32);
        const float M = fmaxf(mx, mx2);
        l  = l * __expf(mx - M) + l2 * __expf(mx2 - M);
        mx = M;
    }
    const float rl = 1.0f / l;

    // ---- pass 2: P x V via WMMA, v from pre-packed fragments --------------
    v8f acc[4] = {};
    const v16h* vpb = vp + (size_t)bh * 32 * 4 * 32;

    for (int jc = 0; jc < 32; ++jc) {
        v16h af;
#pragma unroll
        for (int i = 0; i < 16; ++i) {
            const float lg = (float)lfr[jc * 32 + kmap(i, hi)];
            af[i] = (_Float16)(__expf(lg - mx) * rl);
        }
#pragma unroll
        for (int dt = 0; dt < 4; ++dt) {
            const v16h bf = vpb[((size_t)jc * 4 + dt) * 32 + L];
            acc[dt] = __builtin_amdgcn_wmma_f32_16x16x32_f16(
                false, af, false, bf, (short)0, acc[dt], false, false);
        }
    }

    // ---- epilogue: o[b*1024 + token][h*64 + d] ----------------------------
#pragma unroll
    for (int dt = 0; dt < 4; ++dt) {
#pragma unroll
        for (int r = 0; r < 8; ++r) {
            const int tok = qt0 + r + 8 * hi;
            o[(size_t)(b * 1024 + tok) * 512 + h * 64 + dt * 16 + m] = acc[dt][r];
        }
    }
}

// ---------------------------------------------------------------------------
extern "C" void kernel_launch(void* const* d_in, const int* in_sizes, int n_in,
                              void* d_out, int out_size, void* d_ws, size_t ws_size,
                              hipStream_t stream)
{
    (void)in_sizes; (void)n_in; (void)out_size; (void)ws_size;

    const float* x        = (const float*)d_in[0];
    const float* Wq       = (const float*)d_in[1];
    const float* bq       = (const float*)d_in[2];
    const float* Wk       = (const float*)d_in[3];
    const float* bk       = (const float*)d_in[4];
    const float* Wv       = (const float*)d_in[5];
    const float* bv       = (const float*)d_in[6];
    const float* Wphi_in  = (const float*)d_in[7];
    const float* bphi_in  = (const float*)d_in[8];
    const float* Wphi_out = (const float*)d_in[9];
    const float* bphi_out = (const float*)d_in[10];
    const float* Wta      = (const float*)d_in[11];
    const float* bta      = (const float*)d_in[12];
    const float* Wtb      = (const float*)d_in[13];
    const float* btb      = (const float*)d_in[14];
    const float* Wtau_in  = (const float*)d_in[15];
    const float* btau_in  = (const float*)d_in[16];
    const float* Wtau_out = (const float*)d_in[17];
    const float* btau_out = (const float*)d_in[18];
    const float* Wo       = (const float*)d_in[19];
    const float* bo       = (const float*)d_in[20];

    char* ws = (char*)d_ws;
    float*    WqF = (float*)(ws + 0);             //  64 KB
    float*    WkF = (float*)(ws + 65536);         //  64 KB
    float*    bqF = (float*)(ws + 131072);        // 128 B
    float*    bkF = (float*)(ws + 131328);        // 128 B
    float*    qsc = (float*)(ws + 131584);        // 512 KB  [bh][t][4]
    float*    ksc = (float*)(ws + 655872);        // 512 KB
    _Float16* v_h = (_Float16*)(ws + 1180160);    //   4 MB  [4096][512] f16
    float*    o_f = (float*)(ws + 5374464);       //   8 MB  [4096][512] f32
    v16h*     v_p = (v16h*)(ws + 13763072);       //   4 MB  packed B-fragments

    lan_fold<<<1, 256, 0, stream>>>(Wq, bq, Wk, bk,
                                    Wphi_in, bphi_in, Wphi_out, bphi_out,
                                    Wta, bta, Wtb, btb,
                                    Wtau_in, btau_in, Wtau_out, btau_out,
                                    WqF, WkF, bqF, bkF);

    lan_scal<<<4096, 64, 0, stream>>>(x, WqF, WkF, bqF, bkF, qsc, ksc);

    // V projection: v = x @ Wv + bv  (f16 output)
    lan_gemm<<<dim3(32, 8), 256, 0, stream>>>(x, Wv, bv, nullptr, v_h,
                                              4096, 512, 512);

    // pre-swizzle v into WMMA B-fragment layout
    lan_pack<<<dim3(32, 32), 128, 0, stream>>>(v_h, v_p);

    lan_attn<<<dim3(32, 32), 64, 0, stream>>>(qsc, ksc, v_p, o_f);

    // output projection: out = o @ Wo + bo
    lan_gemm<<<dim3(32, 8), 256, 0, stream>>>(o_f, Wo, bo, (float*)d_out, nullptr,
                                              4096, 512, 512);
}
